// MultiHeadAttentionCombinedQKV_3143916060976
// MI455X (gfx1250) — matrix operand order repaired and checked by reading, not compiled, not run on hardware
//
#include <hip/hip_runtime.h>
#include <hip/hip_bf16.h>
#include <stdint.h>

#define D_INF   1024
#define D_OUTF  1024
#define BATCH   2
#define SEQ     2048
#define NH      16
#define HD      64
#define M_ROWS  (BATCH*SEQ)      // 4096
#define N_QKV   (3*D_OUTF)       // 3072

typedef __attribute__((ext_vector_type(16))) __bf16    v16bf;
typedef __attribute__((ext_vector_type(8)))  float     v8f;
typedef __attribute__((ext_vector_type(4)))  float     f4;
typedef __attribute__((ext_vector_type(4)))  uint32_t  u32x4;

union FragB16 {
    v16bf    v;
    uint32_t u[8];
    u32x4    q[2];
};

__device__ __forceinline__ uint32_t pk2(float lo, float hi) {
    return (__builtin_bit_cast(uint32_t, lo) >> 16) |
           (__builtin_bit_cast(uint32_t, hi) & 0xFFFF0000u);
}
__device__ __forceinline__ uint16_t to_bf16(float x) {
    return (uint16_t)(__builtin_bit_cast(uint32_t, x) >> 16);
}

// A-fragment from bf16 memory: K chunks {0..7} and {16..23} relative to base.
__device__ __forceinline__ void load_bf16_frag_a(FragB16& f, const uint16_t* base) {
    f.q[0] = *(const u32x4*)(base);
    f.q[1] = *(const u32x4*)(base + 16);
}
// B-fragment from bf16 memory: K chunks {0..7} and {16..23} relative to base.
__device__ __forceinline__ void load_bf16_frag16(FragB16& f, const uint16_t* base) {
    f.q[0] = *(const u32x4*)(base);
    f.q[1] = *(const u32x4*)(base + 16);
}

// ---------------------------------------------------------------------------
// Kernel 0: fp32 -> bf16 bulk conversion (8 elements / thread, b128 in/out)
// ---------------------------------------------------------------------------
__global__ __launch_bounds__(256) void cvt_bf16_kernel(
        const float* __restrict__ in, uint16_t* __restrict__ out, int n8) {
    const int i = blockIdx.x * blockDim.x + threadIdx.x;
    if (i >= n8) return;
    const f4* p = (const f4*)in + (size_t)i * 2;
    const f4 a = p[0];
    const f4 b = p[1];
    u32x4 o;
    o.x = pk2(a.x, a.y);
    o.y = pk2(a.z, a.w);
    o.z = pk2(b.x, b.y);
    o.w = pk2(b.z, b.w);
    ((u32x4*)out)[i] = o;
}

// ---------------------------------------------------------------------------
// Kernel 1: QKV = Xb * Wqkvb^T (bf16 WMMA), software-pipelined B-fragments.
// Scatter to per-head Q,K [B,H,S,D] and V^T [B,H,D,S] (bf16).
// One wave -> 16x128 output tile (8 WMMAs / K-step). 8 waves/block.
// ---------------------------------------------------------------------------
__global__ __launch_bounds__(256) void qkv_gemm_kernel(
        const uint16_t* __restrict__ xb, const uint16_t* __restrict__ wb,
        uint16_t* __restrict__ qb, uint16_t* __restrict__ kbuf,
        uint16_t* __restrict__ vT) {
    const int lane = threadIdx.x & 31;
    const int wid  = blockIdx.x * (blockDim.x >> 5) + (threadIdx.x >> 5);
    const int NGRP = N_QKV / 128;           // 24
    const int mt = wid / NGRP;
    const int ng = wid % NGRP;
    if (mt >= M_ROWS / 16) return;          // wave-uniform
    const int m0 = mt * 16, n0 = ng * 128;
    const int half = lane >> 4, l16 = lane & 15;

    const uint16_t* aptr = xb + (size_t)(m0 + l16) * D_INF;
    const uint16_t* bptr = wb + (size_t)(n0 + l16) * D_INF + half * 8;
    v8f acc[8] = {};

    for (int k0 = 0; k0 < D_INF; k0 += 32) {
        FragB16 af;
        load_bf16_frag_a(af, aptr + k0 + half * 8);
        __builtin_prefetch(aptr + k0 + 32, 0, 3);
        FragB16 bf[2];
        load_bf16_frag16(bf[0], bptr + k0);
#pragma unroll
        for (int t = 0; t < 8; ++t) {
            if (t < 7)
                load_bf16_frag16(bf[(t + 1) & 1], bptr + (size_t)(t + 1) * 16 * D_INF + k0);
            acc[t] = __builtin_amdgcn_wmma_f32_16x16x32_bf16(
                false, af.v, false, bf[t & 1].v, (short)0, acc[t], false, false);
        }
    }
#pragma unroll
    for (int t = 0; t < 8; ++t) {
        const int n = n0 + t * 16 + l16;
        const int which = n >> 10;
        const int h = (n >> 6) & (NH - 1);
        const int d = n & (HD - 1);
#pragma unroll
        for (int r = 0; r < 8; ++r) {
            const int m = m0 + half * 8 + r;
            const int b = m >> 11, s = m & (SEQ - 1);
            const uint16_t val = to_bf16(acc[t][r]);
            if (which == 0)      qb  [((size_t)(b * NH + h) * SEQ + s) * HD + d] = val;
            else if (which == 1) kbuf[((size_t)(b * NH + h) * SEQ + s) * HD + d] = val;
            else                 vT  [((size_t)(b * NH + h) * HD + d) * SEQ + s] = val;
        }
    }
}

// ---------------------------------------------------------------------------
// Kernel 2: causal flash attention per (b,h). One wave -> 16 query rows.
// Online softmax over 32-key blocks; scores & PV via bf16 WMMA.
// V-fragment loads are issued before the softmax VALU block to hide latency.
// ---------------------------------------------------------------------------
__global__ __launch_bounds__(256) void attn_kernel(
        const uint16_t* __restrict__ qb, const uint16_t* __restrict__ kbuf,
        const uint16_t* __restrict__ vT, uint16_t* __restrict__ ctx) {
    __shared__ uint16_t plds[8 * 16 * 32];    // 1KB per wave: P tile staging
    const int lane = threadIdx.x & 31;
    const int w = threadIdx.x >> 5;
    const int bh = blockIdx.x >> 4;           // 0..31 = b*16+h
    const int qc = blockIdx.x & 15;
    const int b = bh >> 4, h = bh & 15;
    const int m0 = qc * 128 + w * 16;
    const int half = lane >> 4, l16 = lane & 15;

    const uint16_t* qh = qb   + (size_t)bh * SEQ * HD;
    const uint16_t* kh = kbuf + (size_t)bh * SEQ * HD;
    const uint16_t* vh = vT   + (size_t)bh * HD * SEQ;
    uint16_t* pw = plds + w * 16 * 32;

    FragB16 qf[2];
    {
        const uint16_t* qrow = qh + (size_t)(m0 + l16) * HD;
        load_bf16_frag_a(qf[0], qrow + 0  + half * 8);
        load_bf16_frag_a(qf[1], qrow + 32 + half * 8);
    }
    float mi[8], li[8];
    v8f o[4] = {};
#pragma unroll
    for (int r = 0; r < 8; ++r) { mi[r] = -3.0e38f; li[r] = 0.f; }

    const int jmax = (m0 + 15) >> 5;
    for (int j = 0; j <= jmax; ++j) {
        const int kb0 = j * 32;
        // ---- scores: batch 4 K-fragment loads, then 4 WMMAs ----
        FragB16 kf[4];
#pragma unroll
        for (int c = 0; c < 2; ++c) {
            load_bf16_frag16(kf[2 * c],     kh + (size_t)(kb0 + l16) * HD + c * 32 + half * 8);
            load_bf16_frag16(kf[2 * c + 1], kh + (size_t)(kb0 + 16 + l16) * HD + c * 32 + half * 8);
        }
        v8f s0 = {}, s1 = {};
        s0 = __builtin_amdgcn_wmma_f32_16x16x32_bf16(false, qf[0].v, false, kf[0].v, (short)0, s0, false, false);
        s1 = __builtin_amdgcn_wmma_f32_16x16x32_bf16(false, qf[0].v, false, kf[1].v, (short)0, s1, false, false);
        s0 = __builtin_amdgcn_wmma_f32_16x16x32_bf16(false, qf[1].v, false, kf[2].v, (short)0, s0, false, false);
        s1 = __builtin_amdgcn_wmma_f32_16x16x32_bf16(false, qf[1].v, false, kf[3].v, (short)0, s1, false, false);

        // ---- issue V-fragment loads early; they retire behind softmax ----
        FragB16 vf[4];
#pragma unroll
        for (int t = 0; t < 4; ++t)
            load_bf16_frag16(vf[t], vh + (size_t)(t * 16 + l16) * SEQ + kb0 + half * 8);

        // scale by sqrt(HD)=8 (reference divides by HD**-0.5) + causal mask
        const int qr0 = m0 + half * 8;
        const int key0 = kb0 + l16;
        float p0[8], p1[8], bmax[8];
#pragma unroll
        for (int r = 0; r < 8; ++r) {
            float a  = s0[r] * 8.0f; if (key0 > qr0 + r)      a  = -3.0e38f;
            float bb = s1[r] * 8.0f; if (key0 + 16 > qr0 + r) bb = -3.0e38f;
            p0[r] = a; p1[r] = bb; bmax[r] = fmaxf(a, bb);
        }
#pragma unroll
        for (int off = 1; off < 16; off <<= 1)
#pragma unroll
            for (int r = 0; r < 8; ++r)
                bmax[r] = fmaxf(bmax[r], __shfl_xor(bmax[r], off, 32));
        float corr[8], rs[8];
#pragma unroll
        for (int r = 0; r < 8; ++r) {
            float mnew = fmaxf(mi[r], bmax[r]);
            corr[r] = __expf(mi[r] - mnew);
            mi[r] = mnew;
            p0[r] = __expf(p0[r] - mnew);
            p1[r] = __expf(p1[r] - mnew);
            rs[r] = p0[r] + p1[r];
        }
#pragma unroll
        for (int off = 1; off < 16; off <<= 1)
#pragma unroll
            for (int r = 0; r < 8; ++r)
                rs[r] += __shfl_xor(rs[r], off, 32);
#pragma unroll
        for (int r = 0; r < 8; ++r) li[r] = li[r] * corr[r] + rs[r];
#pragma unroll
        for (int t = 0; t < 4; ++t)
#pragma unroll
            for (int r = 0; r < 8; ++r) o[t][r] *= corr[r];

        // C-layout -> A-layout via LDS (same-wave DS ops are in-order)
#pragma unroll
        for (int r = 0; r < 8; ++r) {
            pw[(half * 8 + r) * 32 + l16]      = to_bf16(p0[r]);
            pw[(half * 8 + r) * 32 + 16 + l16] = to_bf16(p1[r]);
        }
        FragB16 pf;
        {
            const uint16_t* prow = pw + l16 * 32 + half * 8;
            pf.q[0] = *(const u32x4*)(prow);
            pf.q[1] = *(const u32x4*)(prow + 16);
        }
#pragma unroll
        for (int t = 0; t < 4; ++t)
            o[t] = __builtin_amdgcn_wmma_f32_16x16x32_bf16(
                false, pf.v, false, vf[t].v, (short)0, o[t], false, false);
    }
    float inv[8];
#pragma unroll
    for (int r = 0; r < 8; ++r) inv[r] = 1.0f / li[r];
#pragma unroll
    for (int t = 0; t < 4; ++t) {
        const int e = h * HD + t * 16 + l16;
#pragma unroll
        for (int r = 0; r < 8; ++r) {
            const int s = m0 + half * 8 + r;
            ctx[((size_t)b * SEQ + s) * D_OUTF + e] = to_bf16(o[t][r] * inv[r]);
        }
    }
}

// ---------------------------------------------------------------------------
// Kernel 3: OUT = ctx * Wprojb^T + bias (bf16 WMMA, fp32 out),
// software-pipelined B-fragments. One wave -> 16x128 output tile.
// ---------------------------------------------------------------------------
__global__ __launch_bounds__(256) void proj_kernel(
        const uint16_t* __restrict__ ctx, const uint16_t* __restrict__ wb,
        const float* __restrict__ bias, float* __restrict__ out) {
    const int lane = threadIdx.x & 31;
    const int wid  = blockIdx.x * (blockDim.x >> 5) + (threadIdx.x >> 5);
    const int NGRP = D_OUTF / 128;           // 8
    const int mt = wid / NGRP;
    const int ng = wid % NGRP;
    if (mt >= M_ROWS / 16) return;
    const int m0 = mt * 16, n0 = ng * 128;
    const int half = lane >> 4, l16 = lane & 15;

    const uint16_t* arow = ctx + (size_t)(m0 + l16) * D_OUTF;
    const uint16_t* bptr = wb + (size_t)(n0 + l16) * D_OUTF + half * 8;
    v8f acc[8] = {};

    for (int k0 = 0; k0 < D_OUTF; k0 += 32) {
        FragB16 af;
        load_bf16_frag_a(af, arow + k0 + half * 8);
        __builtin_prefetch(arow + k0 + 32, 0, 3);
        FragB16 bf[2];
        load_bf16_frag16(bf[0], bptr + k0);
#pragma unroll
        for (int t = 0; t < 8; ++t) {
            if (t < 7)
                load_bf16_frag16(bf[(t + 1) & 1], bptr + (size_t)(t + 1) * 16 * D_OUTF + k0);
            acc[t] = __builtin_amdgcn_wmma_f32_16x16x32_bf16(
                false, af.v, false, bf[t & 1].v, (short)0, acc[t], false, false);
        }
    }
#pragma unroll
    for (int t = 0; t < 8; ++t) {
        const int n = n0 + t * 16 + l16;
        const float bv = bias[n];
#pragma unroll
        for (int r = 0; r < 8; ++r) {
            const int m = m0 + half * 8 + r;
            out[(size_t)m * D_OUTF + n] = acc[t][r] + bv;
        }
    }
}

extern "C" void kernel_launch(void* const* d_in, const int* in_sizes, int n_in,
                              void* d_out, int out_size, void* d_ws, size_t ws_size,
                              hipStream_t stream) {
    const float* x     = (const float*)d_in[0];
    const float* wqkv  = (const float*)d_in[1];
    const float* wproj = (const float*)d_in[2];
    const float* bproj = (const float*)d_in[3];
    float* out = (float*)d_out;

    // Workspace layout (bf16 / ushort), 40MB total:
    //   [0,   8MB) xb   (bf16 X)    -- dead after qkv_gemm, reused as ctx
    //   [8,  14MB) wqkvb
    //   [14, 16MB) wprojb
    //   [16, 24MB) qb
    //   [24, 32MB) kb
    //   [32, 40MB) vT
    uint16_t* xb     = (uint16_t*)d_ws;
    uint16_t* ctx    = xb;                                   // alias (stream-ordered)
    uint16_t* wqkvb  = xb + (size_t)M_ROWS * D_INF;          // 4096*1024
    uint16_t* wprojb = wqkvb + (size_t)N_QKV * D_INF;        // +3072*1024
    uint16_t* qb     = wprojb + (size_t)D_OUTF * D_OUTF;     // +1024*1024
    uint16_t* kb     = qb + (size_t)M_ROWS * D_OUTF;
    uint16_t* vT     = kb + (size_t)M_ROWS * D_OUTF;

    // fp32 -> bf16 conversions (8 elems/thread)
    cvt_bf16_kernel<<<(M_ROWS * D_INF / 8 + 255) / 256, 256, 0, stream>>>(x, xb, M_ROWS * D_INF / 8);
    cvt_bf16_kernel<<<(N_QKV * D_INF / 8 + 255) / 256, 256, 0, stream>>>(wqkv, wqkvb, N_QKV * D_INF / 8);
    cvt_bf16_kernel<<<(D_OUTF * D_OUTF / 8 + 255) / 256, 256, 0, stream>>>(wproj, wprojb, D_OUTF * D_OUTF / 8);

    // 256 m-tiles * 24 n-groups = 6144 waves / 8 per block
    qkv_gemm_kernel<<<768, 256, 0, stream>>>(xb, wqkvb, qb, kb, vT);
    // 32 (b,h) * 16 q-chunks of 128 rows
    attn_kernel<<<512, 256, 0, stream>>>(qb, kb, vT, ctx);
    // 256 m-tiles * 8 n-groups = 2048 waves / 8 per block
    proj_kernel<<<256, 256, 0, stream>>>(ctx, wprojb, bproj, out);
}
